// GCNModel_10007273800343
// MI455X (gfx1250) — matrix operand order repaired and checked by reading, not compiled, not run on hardware
//
#include <hip/hip_runtime.h>
#include <hip/hip_bf16.h>

// ---------------- problem constants (from reference) ----------------
#define GN   50000
#define GE   800000
#define GIN  256
#define GHID 256
#define GOUT 64
#define HEADS 4
#define DH    64
#define EPS_BN 1e-5f

typedef __attribute__((ext_vector_type(4)))  float    f4;
typedef __attribute__((ext_vector_type(16))) _Float16 v16h;
typedef __attribute__((ext_vector_type(8)))  float    v8f;

// ---------------------------------------------------------------
// B repack: f32 weight matrix [K x Nc] -> f16 fragments in WMMA
// B-operand order.  Packed layout (halfword index):
//   p = ((ks * nTiles + tile) * 32 + lane) * 16 + j
// where lane holds column n = tile*16 + (lane&15) and
// K index = ks*32 + 8*(lane>>4) + (j&7) + 16*(j>>3)  (zero-filled past K).
// Weights are tiny (<=256KB) -> stay resident in the 192MB L2, so
// every wave can read its 32B fragment directly, no LDS staging.
// ---------------------------------------------------------------
__global__ void k_repackB(const float* __restrict__ B, _Float16* __restrict__ Bp,
                          int K, int Nc, long long total) {
    int nTiles = Nc >> 4;
    for (long long p = (long long)blockIdx.x * blockDim.x + threadIdx.x;
         p < total; p += (long long)gridDim.x * blockDim.x) {
        int j    = (int)(p & 15);
        int lane = (int)((p >> 4) & 31);
        long long rest = p >> 9;                 // = ks*nTiles + tile
        int tile = (int)(rest % nTiles);
        int ks   = (int)(rest / nTiles);
        int n  = tile * 16 + (lane & 15);
        int kk = ks * 32 + ((lane & 16) >> 1) + (j & 7) + ((j & 8) << 1);
        float v = (kk < K) ? B[(long)kk * Nc + n] : 0.0f;
        Bp[p] = (_Float16)v;
    }
}

// ---------------------------------------------------------------
// WMMA GEMM:  C[M x Nc] = A[M x K] @ B[K x Nc] (+ bias)
// A: f32 in memory, converted to f16 fragments in-register.
// B: pre-packed f16 fragments (k_repackB) read straight from L2.
// 4 waves / block; each wave owns a 16 x (16*NT) output strip.
// No LDS, no barriers: inner loop is loads + cvt + NT WMMAs.
// Branch-free (clamp + select) so EXEC is all-ones at the WMMAs.
// ---------------------------------------------------------------
template <int NT>
__global__ __launch_bounds__(128) void gemm_wmma_f16(
    const float* __restrict__ A, const _Float16* __restrict__ Bp,
    const float* __restrict__ bias, float* __restrict__ C,
    int M, int K, int Nc)
{
    const int lane = threadIdx.x & 31;
    const int wave = threadIdx.x >> 5;
    const int l15  = lane & 15;
    const int hsel = lane >> 4;                 // 0: lanes 0-15, 1: lanes 16-31
    const int nTiles  = Nc >> 4;
    const int rowBase = (blockIdx.x * 4 + wave) * 16;
    const int colTile = blockIdx.y * NT;
    const int row     = rowBase + l15;
    const int rowLd   = row < M ? row : M - 1;  // clamp: loads always legal
    const float* Arow = A + (long)rowLd * K;
    const int KS = (K + 31) >> 5;

    v8f acc[NT];
#pragma unroll
    for (int t = 0; t < NT; ++t) acc[t] = v8f{};

    for (int ks = 0; ks < KS; ++ks) {
        // ---- A fragment (ISA 16-bit A 16x32 layout) ----
        // lane half 0: K = {k0..k0+7, k0+16..k0+23}; half 1: +8.
        const int  k0  = ks * 32;
        const int  ka  = k0 + hsel * 8;
        const bool ok0 = (ka + 8  <= K);
        const bool ok1 = (ka + 24 <= K);
        f4 q0 = *(const f4*)(Arow + (ok0 ? ka      : 0));
        f4 q1 = *(const f4*)(Arow + (ok0 ? ka + 4  : 0));
        f4 q2 = *(const f4*)(Arow + (ok1 ? ka + 16 : 0));
        f4 q3 = *(const f4*)(Arow + (ok1 ? ka + 20 : 0));
        union { v16h v; _Float16 h[16]; } a;
#pragma unroll
        for (int i = 0; i < 4; ++i) {
            a.h[i]      = (_Float16)(ok0 ? q0[i] : 0.0f);
            a.h[4 + i]  = (_Float16)(ok0 ? q1[i] : 0.0f);
            a.h[8 + i]  = (_Float16)(ok1 ? q2[i] : 0.0f);
            a.h[12 + i] = (_Float16)(ok1 ? q3[i] : 0.0f);
        }
        if (k0 + 32 < K)   // uniform: hint next A tile (global_prefetch_b8)
            __builtin_prefetch(Arow + k0 + 32, 0, 0);

        // ---- NT WMMAs sharing one A fragment; B fragments from L2 ----
        const _Float16* bp =
            Bp + (((long)ks * nTiles + colTile) * 32 + lane) * 16;
#pragma unroll
        for (int t = 0; t < NT; ++t) {
            v16h bfrag = *(const v16h*)(bp + (long)t * 32 * 16);  // 32B vector load
            acc[t] = __builtin_amdgcn_wmma_f32_16x16x32_f16(
                false, a.v, false, bfrag, (short)0, acc[t], false, false);
        }
    }

    // C/D layout: element r -> (row = rowBase + r + 8*hsel, col = base + l15)
#pragma unroll
    for (int t = 0; t < NT; ++t) {
        const int col = (colTile + t) * 16 + l15;
        const float bv = bias ? bias[col] : 0.0f;
#pragma unroll
        for (int r = 0; r < 8; ++r) {
            int rr = rowBase + r + 8 * hsel;
            if (rr < M) C[(long)rr * Nc + col] = acc[t][r] + bv;
        }
    }
}

// ---------------- elementwise / scatter helpers ----------------
#define GS_LOOP(i, total) \
    for (long long i = (long long)blockIdx.x * blockDim.x + threadIdx.x; \
         i < (total); i += (long long)gridDim.x * blockDim.x)

__global__ void k_fill(float* p, float v, long long n) { GS_LOOP(i, n) p[i] = v; }

__global__ void k_degree(const long long* __restrict__ dst, float* __restrict__ deg, long long e) {
    GS_LOOP(i, e) atomicAdd(&deg[(int)dst[i]], 1.0f);
}
__global__ void k_rsqrt_ip(float* p, long long n) { GS_LOOP(i, n) p[i] = rsqrtf(p[i]); }

// out[dst,c] += h[src,c] * dinv[src]*dinv[dst]  (lanes = consecutive c: coalesced)
template <int C>
__global__ void k_gcn_edge(const float* __restrict__ h, const float* __restrict__ dinv,
                           const long long* __restrict__ src, const long long* __restrict__ dst,
                           float* __restrict__ out, long long e) {
    GS_LOOP(t, e * (long long)C) {
        long long ed = t / C; int f = (int)(t & (C - 1));
        int s = (int)src[ed], d = (int)dst[ed];
        atomicAdd(&out[(long)d * C + f], h[(long)s * C + f] * dinv[s] * dinv[d]);
    }
}
// agg += h * dinv^2 + bias  (GCN self-loop term), in place
template <int C>
__global__ void k_gcn_self_bias(float* __restrict__ agg, const float* __restrict__ h,
                                const float* __restrict__ dinv, const float* __restrict__ b,
                                long long n) {
    GS_LOOP(t, n * (long long)C) {
        long long r = t / C; int f = (int)(t & (C - 1));
        float dv = dinv[r];
        agg[t] = agg[t] + h[t] * dv * dv + b[f];
    }
}

// per-column sum / sumsq (one block per column)
__global__ __launch_bounds__(256) void k_bn_stats(const float* __restrict__ x,
                                                  float* __restrict__ stats, int n, int c) {
    __shared__ float ssum[256], ssq[256];
    int col = blockIdx.x, t = threadIdx.x;
    float s = 0.f, q = 0.f;
    for (int r = t; r < n; r += 256) { float v = x[(long)r * c + col]; s += v; q += v * v; }
    ssum[t] = s; ssq[t] = q; __syncthreads();
    for (int o = 128; o > 0; o >>= 1) {
        if (t < o) { ssum[t] += ssum[t + o]; ssq[t] += ssq[t + o]; }
        __syncthreads();
    }
    if (t == 0) { stats[col] = ssum[0]; stats[c + col] = ssq[0]; }
}

// y = leaky_relu( g*(x-mu)*rsqrt(var+eps)+be, 0.01 ) + res
template <int C>
__global__ void k_bn_apply(const float* __restrict__ x, const float* __restrict__ stats,
                           const float* __restrict__ g, const float* __restrict__ be,
                           const float* __restrict__ res, float* __restrict__ y,
                           long long n) {
    GS_LOOP(t, n * (long long)C) {
        int f = (int)(t & (C - 1));
        float mu  = stats[f] / (float)n;
        float var = stats[C + f] / (float)n - mu * mu;
        float v = g[f] * (x[t] - mu) * rsqrtf(var + EPS_BN) + be[f];
        v = v > 0.f ? v : 0.01f * v;
        y[t] = v + res[t];
    }
}

// ---------------- GAT pieces ----------------
__device__ __forceinline__ unsigned f2o(float f) {
    unsigned u = __float_as_uint(f);
    return (u & 0x80000000u) ? ~u : (u | 0x80000000u);
}
__device__ __forceinline__ float o2f(unsigned u) {
    return (u & 0x80000000u) ? __uint_as_float(u & 0x7fffffffu) : __uint_as_float(~u);
}
__device__ __forceinline__ float lrelu02(float v) { return v > 0.f ? v : 0.2f * v; }

__global__ void k_att_logits(const float* __restrict__ hg, const float* __restrict__ a_s,
                             const float* __restrict__ a_d, float* __restrict__ als,
                             float* __restrict__ ald, long long n) {
    GS_LOOP(t, n * HEADS) {
        long long nd = t >> 2; int h = (int)(t & 3);
        const float* hv = hg + nd * (HEADS * DH) + h * DH;
        float s = 0.f, d = 0.f;
        for (int k = 0; k < DH; ++k) { s += hv[k] * a_s[h * DH + k]; d += hv[k] * a_d[h * DH + k]; }
        als[t] = s; ald[t] = d;
    }
}
__global__ void k_gat_init_max(const float* als, const float* ald, unsigned* mu, long long n) {
    GS_LOOP(t, n * HEADS) mu[t] = f2o(lrelu02(als[t] + ald[t]));
}
__global__ void k_gat_edge_max(const float* als, const float* ald,
                               const long long* src, const long long* dst,
                               unsigned* mu, long long e) {
    GS_LOOP(t, e * HEADS) {
        long long ed = t >> 2; int h = (int)(t & 3);
        int s = (int)src[ed], d = (int)dst[ed];
        atomicMax(&mu[(long)d * HEADS + h],
                  f2o(lrelu02(als[(long)s * HEADS + h] + ald[(long)d * HEADS + h])));
    }
}
__global__ void k_gat_init_acc(const float* als, const float* ald, const float* __restrict__ hg,
                               float* mbuf, float* denom, float* __restrict__ numer, long long n) {
    GS_LOOP(t, n * HEADS) {
        long long nd = t >> 2; int h = (int)(t & 3);
        float m = o2f(__float_as_uint(mbuf[t]));
        mbuf[t] = m;
        float es = __expf(lrelu02(als[t] + ald[t]) - m);
        denom[t] = es;
        const float* hv = hg + nd * (HEADS * DH) + h * DH;
        float* nv = numer + t * DH;
        for (int k = 0; k < DH; ++k) nv[k] = es * hv[k];
    }
}
__global__ void k_gat_edge_acc(const float* als, const float* ald, const float* __restrict__ hg,
                               const long long* src, const long long* dst,
                               const float* mbuf, float* denom, float* __restrict__ numer,
                               long long e) {
    GS_LOOP(t, e * HEADS) {
        long long ed = t >> 2; int h = (int)(t & 3);
        int s = (int)src[ed], d = (int)dst[ed];
        float ee = __expf(lrelu02(als[(long)s * HEADS + h] + ald[(long)d * HEADS + h])
                          - mbuf[(long)d * HEADS + h]);
        atomicAdd(&denom[(long)d * HEADS + h], ee);
        const float* hv = hg + (long)s * (HEADS * DH) + h * DH;
        float* nv = numer + ((long)d * HEADS + h) * DH;
        for (int k = 0; k < DH; ++k) atomicAdd(&nv[k], ee * hv[k]);
    }
}
__global__ void k_gat_final(const float* __restrict__ numer, const float* __restrict__ denom,
                            const float* __restrict__ bg, float* __restrict__ out, long long n) {
    GS_LOOP(t, n * DH) {
        long long nd = t >> 6; int k = (int)(t & 63);
        float s = 0.f;
        for (int h = 0; h < HEADS; ++h)
            s += numer[(nd * HEADS + h) * DH + k] / denom[nd * HEADS + h];
        out[t] = s * (1.0f / HEADS) + bg[k];
    }
}

// ---------------- driver ----------------
extern "C" void kernel_launch(void* const* d_in, const int* in_sizes, int n_in,
                              void* d_out, int out_size, void* d_ws, size_t ws_size,
                              hipStream_t stream) {
    (void)in_sizes; (void)n_in; (void)out_size; (void)ws_size;
    const float*     x  = (const float*)d_in[0];
    const long long* ei = (const long long*)d_in[1];
    const float *W1 = (const float*)d_in[2],  *b1 = (const float*)d_in[3];
    const float *g1 = (const float*)d_in[4],  *be1 = (const float*)d_in[5];
    const float *Wg = (const float*)d_in[6],  *a_s = (const float*)d_in[7];
    const float *a_d = (const float*)d_in[8], *bg = (const float*)d_in[9];
    const float *g2 = (const float*)d_in[10], *be2 = (const float*)d_in[11];
    const float *W3 = (const float*)d_in[12], *b3 = (const float*)d_in[13];
    const float *g3 = (const float*)d_in[14], *be3 = (const float*)d_in[15];
    const float *W4 = (const float*)d_in[16], *b4 = (const float*)d_in[17];
    const float *r1W = (const float*)d_in[18], *r1b = (const float*)d_in[19];
    const float *r2W = (const float*)d_in[20], *r2b = (const float*)d_in[21];
    const float *r3W = (const float*)d_in[22], *r3b = (const float*)d_in[23];
    const float *pW = (const float*)d_in[24], *pb = (const float*)d_in[25];
    const long long *src = ei, *dst = ei + GE;
    float* out = (float*)d_out;

    // workspace carving (floats)
    float* W = (float*)d_ws; size_t off = 0;
    auto alloc = [&](size_t n) { float* p = W + off; off += (n + 15) & ~(size_t)15; return p; };
    float* h256  = alloc((size_t)GN * 256);
    float* agg   = alloc((size_t)GN * 256);
    float* r256  = alloc((size_t)GN * 256);
    float* x2    = alloc((size_t)GN * 256);
    float* hg    = alloc((size_t)GN * 256);
    float* numer = alloc((size_t)GN * 256);
    float* g64   = alloc((size_t)GN * 64);
    float* r64   = alloc((size_t)GN * 64);
    float* x3    = alloc((size_t)GN * 64);
    float* h16   = alloc((size_t)GN * 16);
    float* agg16 = alloc((size_t)GN * 16);
    float* r16   = alloc((size_t)GN * 16);
    float* x4    = alloc((size_t)GN * 16);
    float* h64b  = alloc((size_t)GN * 64);
    float* agg64 = alloc((size_t)GN * 64);
    float* dinv  = alloc(GN);
    float* stats = alloc(512);
    float* als   = alloc((size_t)GN * HEADS);
    float* ald   = alloc((size_t)GN * HEADS);
    float* mbuf  = alloc((size_t)GN * HEADS);
    float* denom = alloc((size_t)GN * HEADS);
    _Float16* Bpack = (_Float16*)alloc(40960);   // max packed B: 8 ksteps*16 tiles*512 h

    auto nb = [](long long total) {
        long long b = (total + 255) / 256; return (int)(b > (1 << 20) ? (1 << 20) : b);
    };
    // repack weight into fragment order, then run WMMA GEMM reading it from L2
    auto gemm = [&](const float* A, const float* B, const float* bias, float* Co,
                    int M, int K, int Nc) {
        int KS = (K + 31) / 32;
        long long total = (long long)KS * (Nc / 16) * 512;
        k_repackB<<<nb(total), 256, 0, stream>>>(B, Bpack, K, Nc, total);
        if (Nc % 64 == 0) {
            dim3 g((M + 63) / 64, Nc / 64);
            gemm_wmma_f16<4><<<g, 128, 0, stream>>>(A, Bpack, bias, Co, M, K, Nc);
        } else {
            dim3 g((M + 63) / 64, Nc / 16);
            gemm_wmma_f16<1><<<g, 128, 0, stream>>>(A, Bpack, bias, Co, M, K, Nc);
        }
    };

    // degrees -> dinv
    k_fill<<<nb(GN), 256, 0, stream>>>(dinv, 1.0f, GN);   // +1 self-loop
    k_degree<<<nb(GE), 256, 0, stream>>>(dst, dinv, GE);
    k_rsqrt_ip<<<nb(GN), 256, 0, stream>>>(dinv, GN);

    // ---- GCN layer 1 (256 -> 256) ----
    gemm(x, W1, nullptr, h256, GN, GIN, GHID);
    k_fill<<<nb((long long)GN * 256), 256, 0, stream>>>(agg, 0.0f, (long long)GN * 256);
    k_gcn_edge<256><<<nb((long long)GE * 256), 256, 0, stream>>>(h256, dinv, src, dst, agg, GE);
    k_gcn_self_bias<256><<<nb((long long)GN * 256), 256, 0, stream>>>(agg, h256, dinv, b1, GN);
    k_bn_stats<<<256, 256, 0, stream>>>(agg, stats, GN, 256);
    gemm(x, r1W, r1b, r256, GN, GIN, GHID);
    k_bn_apply<256><<<nb((long long)GN * 256), 256, 0, stream>>>(agg, stats, g1, be1, r256, x2, GN);

    // ---- GAT layer (256 -> 64, 4 heads, mean) ----
    gemm(x2, Wg, nullptr, hg, GN, GHID, HEADS * DH);
    k_att_logits<<<nb((long long)GN * HEADS), 256, 0, stream>>>(hg, a_s, a_d, als, ald, GN);
    k_gat_init_max<<<nb((long long)GN * HEADS), 256, 0, stream>>>(als, ald, (unsigned*)mbuf, GN);
    k_gat_edge_max<<<nb((long long)GE * HEADS), 256, 0, stream>>>(als, ald, src, dst, (unsigned*)mbuf, GE);
    k_gat_init_acc<<<nb((long long)GN * HEADS), 256, 0, stream>>>(als, ald, hg, mbuf, denom, numer, GN);
    k_gat_edge_acc<<<nb((long long)GE * HEADS), 256, 0, stream>>>(als, ald, hg, src, dst, mbuf, denom, numer, GE);
    k_gat_final<<<nb((long long)GN * 64), 256, 0, stream>>>(numer, denom, bg, g64, GN);
    k_bn_stats<<<64, 256, 0, stream>>>(g64, stats, GN, 64);
    gemm(x2, r2W, r2b, r64, GN, GHID, 64);
    k_bn_apply<64><<<nb((long long)GN * 64), 256, 0, stream>>>(g64, stats, g2, be2, r64, x3, GN);

    // ---- GCN layer 3 (64 -> 16) ----
    gemm(x3, W3, nullptr, h16, GN, 64, 16);
    k_fill<<<nb((long long)GN * 16), 256, 0, stream>>>(agg16, 0.0f, (long long)GN * 16);
    k_gcn_edge<16><<<nb((long long)GE * 16), 256, 0, stream>>>(h16, dinv, src, dst, agg16, GE);
    k_gcn_self_bias<16><<<nb((long long)GN * 16), 256, 0, stream>>>(agg16, h16, dinv, b3, GN);
    k_bn_stats<<<16, 256, 0, stream>>>(agg16, stats, GN, 16);
    gemm(x3, r3W, r3b, r16, GN, 64, 16);
    k_bn_apply<16><<<nb((long long)GN * 16), 256, 0, stream>>>(agg16, stats, g3, be3, r16, x4, GN);

    // ---- GCN layer 4 (16 -> 64), K=16 zero-padded at repack ----
    gemm(x4, W4, nullptr, h64b, GN, 16, 64);
    k_fill<<<nb((long long)GN * 64), 256, 0, stream>>>(agg64, 0.0f, (long long)GN * 64);
    k_gcn_edge<64><<<nb((long long)GE * 64), 256, 0, stream>>>(h64b, dinv, src, dst, agg64, GE);
    k_gcn_self_bias<64><<<nb((long long)GN * 64), 256, 0, stream>>>(agg64, h64b, dinv, b4, GN);

    // ---- output projection ----
    gemm(agg64, pW, pb, out, GN, 64, 64);
}
